// TransolverBlock_37915971289756
// MI455X (gfx1250) — matrix operand order repaired and checked by reading, not compiled, not run on hardware
//
#include <hip/hip_runtime.h>
#include <hip/hip_bf16.h>

typedef __attribute__((ext_vector_type(16))) __bf16 bf16x16;
typedef __attribute__((ext_vector_type(8)))  __bf16 bf16x8;
typedef __attribute__((ext_vector_type(8)))  float  f32x8;

#define BB   2
#define NTOK 32768
#define CDIM 256
#define HH   8
#define MM   64
#define DD   32
#define RTOK (BB * NTOK)   // 65536 rows

// ---------------------------------------------------------------------------
// Weight transpose + f32 -> bf16 convert:  W[K][Nc] -> Wt[Nc][K]
// ---------------------------------------------------------------------------
__global__ void wt_to_bf16_t(const float* __restrict__ W, __bf16* __restrict__ Wt,
                             int K, int Nc) {
  int idx = blockIdx.x * 256 + threadIdx.x;
  if (idx >= K * Nc) return;
  int k = idx / Nc, n = idx - k * Nc;
  Wt[(size_t)n * K + k] = (__bf16)W[idx];
}

// ---------------------------------------------------------------------------
// LayerNorm over C=256, one wave32 per row, output bf16
// ---------------------------------------------------------------------------
__global__ __launch_bounds__(256)
void layernorm_bf16(const float* __restrict__ x, const float* __restrict__ g,
                    const float* __restrict__ b, __bf16* __restrict__ h) {
  int row  = blockIdx.x * 8 + (threadIdx.x >> 5);
  int lane = threadIdx.x & 31;
  const float* xr = x + (size_t)row * CDIM;
  float v[8];
  float s = 0.f, sq = 0.f;
#pragma unroll
  for (int j = 0; j < 8; ++j) {
    v[j] = xr[lane + j * 32];
    s += v[j]; sq += v[j] * v[j];
  }
#pragma unroll
  for (int off = 16; off > 0; off >>= 1) {
    s  += __shfl_xor(s,  off, 32);
    sq += __shfl_xor(sq, off, 32);
  }
  float mu   = s * (1.f / CDIM);
  float var  = sq * (1.f / CDIM) - mu * mu;
  float rinv = rsqrtf(var + 1e-5f);
  __bf16* hr = h + (size_t)row * CDIM;
#pragma unroll
  for (int j = 0; j < 8; ++j) {
    int c = lane + j * 32;
    hr[c] = (__bf16)((v[j] - mu) * rinv * g[c] + b[c]);
  }
}

// ---------------------------------------------------------------------------
// bf16 WMMA GEMM:  out[row][col] = A[row][.] @ Bt[col][.] + bias (+res / gelu)
//   A  : row-major bf16 [Nrows x K]
//   Bt : row-major bf16 [Ncols x K]   (weight pre-transposed)
// block = 128 threads = 4 waves; wave tile = 16 rows x 64 cols
// MODE 0: f32 out + bias   MODE 1: f32 out + bias + residual
// MODE 2: bf16 out, gelu(v + bias)
// ---------------------------------------------------------------------------
template<int MODE>
__global__ __launch_bounds__(128)
void gemm_bf16_wmma(const __bf16* __restrict__ A, const __bf16* __restrict__ Bt,
                    const float* __restrict__ bias, const float* __restrict__ res,
                    void* __restrict__ outp, int K, int Ncols) {
  int lane  = threadIdx.x & 31;
  int wave  = threadIdx.x >> 5;
  int nlane = lane & 15;
  int khA   = (lane >> 4) * 16;   // A K-half offset
  int khB   = (lane >> 4) * 16;   // B K-half offset
  int rowA  = blockIdx.y * 64 + wave * 16 + nlane;
  int colBase = blockIdx.x * 64;

  f32x8 acc[4] = {};

  for (int k0 = 0; k0 < K; k0 += 32) {
    union { bf16x16 v; bf16x8 h[2]; } a;
    const __bf16* ap = A + (size_t)rowA * K + (k0 + khA);
    a.h[0] = *(const bf16x8*)ap;           // K = kh..kh+7
    a.h[1] = *(const bf16x8*)(ap + 8);     // K = kh+8..kh+15
#pragma unroll
    for (int c = 0; c < 4; ++c) {
      const __bf16* bp = Bt + (size_t)(colBase + c * 16 + nlane) * K + (k0 + khB);
      bf16x16 bv = *(const bf16x16*)bp;    // 16 contiguous K values
      acc[c] = __builtin_amdgcn_wmma_f32_16x16x32_bf16(
          false, a.v, false, bv, (short)0, acc[c], false, false);
    }
  }

  int rowO = blockIdx.y * 64 + wave * 16 + (lane >> 4) * 8;
#pragma unroll
  for (int c = 0; c < 4; ++c) {
    int col = colBase + c * 16 + nlane;
    float bc = bias[col];
#pragma unroll
    for (int r = 0; r < 8; ++r) {
      size_t oidx = (size_t)(rowO + r) * Ncols + col;
      float v = acc[c][r] + bc;
      if (MODE == 1) v += res[oidx];
      if (MODE == 2) {
        float ge = 0.5f * v * (1.0f + erff(v * 0.70710678118654752f));
        ((__bf16*)outp)[oidx] = (__bf16)ge;
      } else {
        ((float*)outp)[oidx] = v;
      }
    }
  }
}

// ---------------------------------------------------------------------------
// Slice routing: per (b,h) chunk of 128 tokens:
//   logits = x_mid @ Wsl + bsl, / clip(temp,0.1); softmax over M
//   write sw to global; partial sums of (sw^T @ fx) and sum_n sw to `part`
// part layout per (bh,chunk): [2048] stoken partial (m*32+d), [64] snorm partial
// ---------------------------------------------------------------------------
__global__ __launch_bounds__(256)
void slice_stats(const float* __restrict__ fx, const float* __restrict__ xm,
                 const float* __restrict__ Wsl, const float* __restrict__ bsl,
                 const float* __restrict__ temp, float* __restrict__ sw_g,
                 float* __restrict__ part) {
  __shared__ float wsl_s[DD][MM];
  __shared__ float bsl_s[MM];
  __shared__ float sw_s[128][MM + 1];
  __shared__ float fx_s[128][DD];
  int tid = threadIdx.x;
  int chunk = blockIdx.x, h = blockIdx.y, b = blockIdx.z;
  int bh = b * HH + h;
  int n0 = chunk * 128;

#pragma unroll
  for (int i = 0; i < 8; ++i) {          // stage Wsl (32x64)
    int p = tid + i * 256;
    wsl_s[p >> 6][p & 63] = Wsl[p];
  }
  if (tid < MM) bsl_s[tid] = bsl[tid];
#pragma unroll
  for (int i = 0; i < 16; ++i) {         // stage fx tile (128x32)
    int p = tid + i * 256;
    int t = p >> 5, d = p & 31;
    fx_s[t][d] = fx[(size_t)(b * NTOK + n0 + t) * CDIM + h * DD + d];
  }
  __syncthreads();

  if (tid < 128) {
    int t = tid, n = n0 + t;
    float xd[DD];
    const float* xr = xm + (size_t)(b * NTOK + n) * CDIM + h * DD;
#pragma unroll
    for (int d = 0; d < DD; ++d) xd[d] = xr[d];
    float itc = 1.0f / fmaxf(temp[h], 0.1f);
    float mx = -3.0e38f;
    for (int m = 0; m < MM; ++m) {
      float a = bsl_s[m];
#pragma unroll
      for (int d = 0; d < DD; ++d) a += xd[d] * wsl_s[d][m];
      a *= itc;
      sw_s[t][m] = a;
      mx = fmaxf(mx, a);
    }
    float ssum = 0.f;
    for (int m = 0; m < MM; ++m) {
      float e = __expf(sw_s[t][m] - mx);
      sw_s[t][m] = e; ssum += e;
    }
    float rs = 1.0f / ssum;
    float* swr = sw_g + ((size_t)bh * NTOK + n) * MM;
    for (int m = 0; m < MM; ++m) {
      float v = sw_s[t][m] * rs;
      sw_s[t][m] = v;
      swr[m] = v;
    }
  }
  __syncthreads();

  float* pp = part + ((size_t)bh * (NTOK / 128) + chunk) * 2112;
#pragma unroll
  for (int i = 0; i < 8; ++i) {          // 2048 = M*D partial outputs
    int p = tid + i * 256;
    int m = p >> 5, d = p & 31;
    float acc = 0.f;
    for (int t = 0; t < 128; ++t) acc += sw_s[t][m] * fx_s[t][d];
    pp[p] = acc;
  }
  if (tid < MM) {
    float s = 0.f;
    for (int t = 0; t < 128; ++t) s += sw_s[t][tid];
    pp[2048 + tid] = s;
  }
}

// Fixed-order (deterministic) reduction of per-chunk partials
__global__ void reduce_slices(const float* __restrict__ part,
                              float* __restrict__ stoken, float* __restrict__ snorm) {
  int idx = blockIdx.x * 256 + threadIdx.x;
  const int TOTAL = BB * HH * 2112;
  if (idx >= TOTAL) return;
  int bh = idx / 2112, p = idx - bh * 2112;
  const float* src = part + ((size_t)bh * (NTOK / 128)) * 2112 + p;
  float acc = 0.f;
  for (int c = 0; c < NTOK / 128; ++c) acc += src[(size_t)c * 2112];
  if (p < 2048) stoken[(size_t)bh * 2048 + p] = acc;
  else          snorm[(size_t)bh * MM + (p - 2048)] = acc;
}

// ---------------------------------------------------------------------------
// Tiny attention over M=64 slices per (b,h): one block per head
// ---------------------------------------------------------------------------
__global__ __launch_bounds__(256)
void small_attn(const float* __restrict__ stoken, const float* __restrict__ snorm,
                const float* __restrict__ Wq, const float* __restrict__ Wk,
                const float* __restrict__ Wv, float* __restrict__ oslice) {
  __shared__ float st[MM][DD + 1];
  __shared__ float qs[MM][DD + 1];
  __shared__ float ks[MM][DD + 1];
  __shared__ float vs[MM][DD + 1];
  __shared__ float at[MM][MM + 1];
  int tid = threadIdx.x;
  int bh = blockIdx.y * HH + blockIdx.x;
  const float* stn = stoken + (size_t)bh * (MM * DD);
  const float* sn  = snorm  + (size_t)bh * MM;
#pragma unroll
  for (int i = 0; i < 8; ++i) {
    int p = tid + i * 256; int m = p >> 5, d = p & 31;
    st[m][d] = stn[p] / fmaxf(sn[m], 1e-5f);
  }
  __syncthreads();
#pragma unroll
  for (int i = 0; i < 8; ++i) {
    int p = tid + i * 256; int m = p >> 5, dj = p & 31;
    float aq = 0.f, ak = 0.f, av = 0.f;
#pragma unroll
    for (int d = 0; d < DD; ++d) {
      float s = st[m][d];
      aq += s * Wq[d * DD + dj];
      ak += s * Wk[d * DD + dj];
      av += s * Wv[d * DD + dj];
    }
    qs[m][dj] = aq; ks[m][dj] = ak; vs[m][dj] = av;
  }
  __syncthreads();
  if (tid < MM) {
    int m = tid;
    const float scale = 0.17677669529663689f;   // 32^-0.5
    float mx = -3.0e38f;
    for (int gq = 0; gq < MM; ++gq) {
      float acc = 0.f;
#pragma unroll
      for (int d = 0; d < DD; ++d) acc += qs[m][d] * ks[gq][d];
      acc *= scale;
      at[m][gq] = acc; mx = fmaxf(mx, acc);
    }
    float ssum = 0.f;
    for (int gq = 0; gq < MM; ++gq) {
      float e = __expf(at[m][gq] - mx);
      at[m][gq] = e; ssum += e;
    }
    float rs = 1.f / ssum;
    for (int gq = 0; gq < MM; ++gq) at[m][gq] *= rs;
  }
  __syncthreads();
  float* osp = oslice + (size_t)bh * (MM * DD);
#pragma unroll
  for (int i = 0; i < 8; ++i) {
    int p = tid + i * 256; int m = p >> 5, d = p & 31;
    float acc = 0.f;
    for (int gq = 0; gq < MM; ++gq) acc += at[m][gq] * vs[gq][d];
    osp[p] = acc;
  }
}

// ---------------------------------------------------------------------------
// Scatter slices back to tokens: out[n][d] = sum_m oslice[m][d] * sw[n][m]
// writes bf16 (B,N,C) layout ready for Wo GEMM; 64 tokens per block
// ---------------------------------------------------------------------------
__global__ __launch_bounds__(256)
void scatter_tokens(const float* __restrict__ oslice, const float* __restrict__ sw_g,
                    __bf16* __restrict__ attn_out) {
  __shared__ float os_s[MM][DD + 1];
  __shared__ float sw_s[64][MM + 1];
  int tid = threadIdx.x;
  int chunk = blockIdx.x, h = blockIdx.y, b = blockIdx.z;
  int bh = b * HH + h;
  int n0 = chunk * 64;
  const float* osp = oslice + (size_t)bh * (MM * DD);
#pragma unroll
  for (int i = 0; i < 8; ++i) {
    int p = tid + i * 256; os_s[p >> 5][p & 31] = osp[p];
  }
#pragma unroll
  for (int i = 0; i < 16; ++i) {
    int p = tid + i * 256; int t = p >> 6, m = p & 63;
    sw_s[t][m] = sw_g[((size_t)bh * NTOK + n0 + t) * MM + m];
  }
  __syncthreads();
#pragma unroll
  for (int i = 0; i < 8; ++i) {
    int p = tid + i * 256; int t = p >> 5, d = p & 31;
    float acc = 0.f;
    for (int m = 0; m < MM; ++m) acc += sw_s[t][m] * os_s[m][d];
    attn_out[(size_t)(b * NTOK + n0 + t) * CDIM + h * DD + d] = (__bf16)acc;
  }
}

// ---------------------------------------------------------------------------
extern "C" void kernel_launch(void* const* d_in, const int* in_sizes, int n_in,
                              void* d_out, int out_size, void* d_ws, size_t ws_size,
                              hipStream_t stream) {
  const float* x    = (const float*)d_in[0];
  const float* ln1g = (const float*)d_in[1];
  const float* ln1b = (const float*)d_in[2];
  const float* Wfx  = (const float*)d_in[3];
  const float* bfx  = (const float*)d_in[4];
  const float* Wx   = (const float*)d_in[5];
  const float* bx   = (const float*)d_in[6];
  const float* Wsl  = (const float*)d_in[7];
  const float* bsl  = (const float*)d_in[8];
  const float* temp = (const float*)d_in[9];
  const float* Wq   = (const float*)d_in[10];
  const float* Wk   = (const float*)d_in[11];
  const float* Wv   = (const float*)d_in[12];
  const float* Wo   = (const float*)d_in[13];
  const float* bo   = (const float*)d_in[14];
  const float* ln2g = (const float*)d_in[15];
  const float* ln2b = (const float*)d_in[16];
  const float* W1   = (const float*)d_in[17];
  const float* b1   = (const float*)d_in[18];
  const float* W2   = (const float*)d_in[19];
  const float* b2   = (const float*)d_in[20];

  char* ws = (char*)d_ws;
  // workspace layout (aliased regions are used at disjoint pipeline stages)
  __bf16* h_bf    = (__bf16*)(ws + 0);                 // 32 MB
  float*  fx_f    = (float*) (ws + (32ull  << 20));    // 64 MB
  float*  xm_f    = (float*) (ws + (96ull  << 20));    // 64 MB
  float*  sw_f    = (float*) (ws + (160ull << 20));    // 128 MB
  float*  part_f  = (float*) (ws + (288ull << 20));    // 33 MB (dead before x1)
  float*  x1_f    = (float*) (ws + (288ull << 20));    // 64 MB
  float*  stok_f  = (float*) (ws + (352ull << 20));    // 128 KB
  float*  snorm_f = (float*) (ws + (352ull << 20) + (128ull << 10)); // 4 KB
  float*  osl_f   = (float*) (ws + (352ull << 20) + (132ull << 10)); // 128 KB
  char*   wtb     = ws + (353ull << 20);
  __bf16* WfxT = (__bf16*)(wtb + 0);
  __bf16* WxT  = (__bf16*)(wtb + (128ull << 10));
  __bf16* WoT  = (__bf16*)(wtb + (256ull << 10));
  __bf16* W1T  = (__bf16*)(wtb + (384ull << 10));      // 512 KB
  __bf16* W2T  = (__bf16*)(wtb + (896ull << 10));      // 512 KB
  __bf16* attnout_bf = (__bf16*)fx_f;                  // reuse fx region
  __bf16* h2_bf      = (__bf16*)xm_f;                  // reuse xm region
  __bf16* mid_bf     = (__bf16*)sw_f;                  // reuse sw region

  // 1) weight transposes (bf16)
  wt_to_bf16_t<<<(256 * 256 + 255) / 256, 256, 0, stream>>>(Wfx, WfxT, 256, 256);
  wt_to_bf16_t<<<(256 * 256 + 255) / 256, 256, 0, stream>>>(Wx,  WxT,  256, 256);
  wt_to_bf16_t<<<(256 * 256 + 255) / 256, 256, 0, stream>>>(Wo,  WoT,  256, 256);
  wt_to_bf16_t<<<(256 * 1024 + 255) / 256, 256, 0, stream>>>(W1, W1T, 256, 1024);
  wt_to_bf16_t<<<(1024 * 256 + 255) / 256, 256, 0, stream>>>(W2, W2T, 1024, 256);

  // 2) LN1 -> bf16
  layernorm_bf16<<<RTOK / 8, 256, 0, stream>>>(x, ln1g, ln1b, h_bf);

  // 3) projections (WMMA)
  dim3 g256(256 / 64, RTOK / 64);
  gemm_bf16_wmma<0><<<g256, 128, 0, stream>>>(h_bf, WfxT, bfx, nullptr, fx_f, 256, 256);
  gemm_bf16_wmma<0><<<g256, 128, 0, stream>>>(h_bf, WxT,  bx,  nullptr, xm_f, 256, 256);

  // 4) slice routing (softmax + deterministic partial reduction)
  slice_stats<<<dim3(NTOK / 128, HH, BB), 256, 0, stream>>>(fx_f, xm_f, Wsl, bsl, temp,
                                                            sw_f, part_f);
  reduce_slices<<<(BB * HH * 2112 + 255) / 256, 256, 0, stream>>>(part_f, stok_f, snorm_f);

  // 5) tiny attention over slices
  small_attn<<<dim3(HH, BB), 256, 0, stream>>>(stok_f, snorm_f, Wq, Wk, Wv, osl_f);

  // 6) scatter back to tokens -> bf16
  scatter_tokens<<<dim3(NTOK / 64, HH, BB), 256, 0, stream>>>(osl_f, sw_f, attnout_bf);

  // 7) output projection + residual -> x1
  gemm_bf16_wmma<1><<<g256, 128, 0, stream>>>(attnout_bf, WoT, bo, x, x1_f, 256, 256);

  // 8) LN2 -> bf16
  layernorm_bf16<<<RTOK / 8, 256, 0, stream>>>(x1_f, ln2g, ln2b, h2_bf);

  // 9) FFN: gelu(h2 @ W1 + b1) -> bf16, then @ W2 + b2 + x1 -> d_out
  dim3 g1024(1024 / 64, RTOK / 64);
  gemm_bf16_wmma<2><<<g1024, 128, 0, stream>>>(h2_bf, W1T, b1, nullptr, mid_bf, 256, 1024);
  gemm_bf16_wmma<1><<<g256, 128, 0, stream>>>(mid_bf, W2T, b2, x1_f, (float*)d_out, 1024, 256);
}
